// GRUNet_25039659335813
// MI455X (gfx1250) — matrix operand order repaired and checked by reading, not compiled, not run on hardware
//
#include <hip/hip_runtime.h>
#include <hip/hip_bf16.h>

// ---------------------------------------------------------------------------
// GRUNet on MI455X (gfx1250): bf16 WMMA for all matmuls, f32 state/gates,
// async global->LDS staging of the xg panel overlapped with the recurrence
// WMMAs.   B=128, T=512, I=64, H=256, G=3H=768, O=2
// ---------------------------------------------------------------------------

#define BB 128
#define TT 512
#define II 64
#define HH 256
#define GG 768   // 3*H

typedef __bf16  v16bf __attribute__((ext_vector_type(16)));
typedef float   v8f   __attribute__((ext_vector_type(8)));

__device__ __forceinline__ unsigned short f2bf(float f) {
    union { float f; unsigned int u; } v; v.f = f;
    unsigned int u = v.u;
    u += 0x7fffu + ((u >> 16) & 1u);          // round-to-nearest-even
    return (unsigned short)(u >> 16);
}

// A-fragment lane mapping used here for 16-bit A 16x32 (contiguous k, not the ISA table order):
//   kk in [0,32): half = kk>>4 ; e = kk&15 ; lane = half*16 + m
__device__ __forceinline__ void a_map(int kk, int& half, int& e) {
    half = kk >> 4;
    e    = kk & 15;
}

// ---------------------------------------------------------------------------
// Kernel 1: pack f32 weight [G rows, K cols] (row = output column of GEMM)
// into B-fragment-linear bf16: frag(ct,kt) = 32 lanes x 32B, lane=half*16+n,
// element e holds W[ct*16+n, kt*32 + 16*half + e].
// ---------------------------------------------------------------------------
__global__ void gru_pack_b(const float* __restrict__ w,
                           unsigned short* __restrict__ dst, int K) {
    int idx = blockIdx.x * blockDim.x + threadIdx.x;
    int KT = K >> 5;
    if (idx >= GG * K) return;
    int e    = idx & 15;
    int lane = (idx >> 4) & 31;
    int kt   = (idx >> 9) % KT;
    int ct   = idx / (512 * KT);
    int half = lane >> 4, n = lane & 15;
    dst[idx] = f2bf(w[(ct * 16 + n) * K + (kt * 32 + half * 16 + e)]);
}

// ---------------------------------------------------------------------------
// Kernel 2: input projection GEMM  xg[t,b,:] = act[row=b*T+t,:] @ W^T + bias.
// K is a template parameter so all fragment addresses fold to base+imm.
// 64 rows x 768 cols per block, 8 waves; wave w owns col tiles [6w,6w+6).
// ---------------------------------------------------------------------------
template<int K, bool ABF16>
__global__ __launch_bounds__(256)
void gru_inproj(const void* __restrict__ Araw,
                const unsigned short* __restrict__ Wp,
                const float* __restrict__ bias,
                float* __restrict__ xg) {
    constexpr int KT = K / 32;
    __shared__ __align__(32) unsigned short afrag[4 * KT * 512];
    const int tid  = threadIdx.x;
    const int row0 = blockIdx.x * 64;

    // Stage A tile (64 x K) into A-fragment layout
    for (int idx = tid; idx < 64 * K; idx += 256) {
        int row = idx / K, k = idx - row * K;
        unsigned short us;
        if (ABF16) us = ((const unsigned short*)Araw)[(size_t)(row0 + row) * K + k];
        else       us = f2bf(((const float*)Araw)[(size_t)(row0 + row) * K + k]);
        int rt = row >> 4, m = row & 15;
        int kt = k >> 5,  kk = k & 31;
        int ha, e; a_map(kk, ha, e);
        afrag[(((rt * KT + kt) * 32) + ha * 16 + m) * 16 + e] = us;
    }
    __syncthreads();

    const int w = tid >> 5, lane = tid & 31;
    const int half = lane >> 4, n = lane & 15;
    const char* ab = (const char*)afrag + lane * 32;                     // + (rt*KT+kt)*1024
    const char* wb = (const char*)Wp + (size_t)w * 6 * KT * 1024 + lane * 32; // + (c*KT+kt)*1024

    for (int rt = 0; rt < 4; ++rt) {
        v8f acc[6] = {};
#pragma unroll
        for (int kt = 0; kt < KT; ++kt) {
            v16bf a = *(const v16bf*)(ab + (rt * KT + kt) * 1024);
#pragma unroll
            for (int c = 0; c < 6; ++c) {
                v16bf bb = *(const v16bf*)(wb + (c * KT + kt) * 1024);
                acc[c] = __builtin_amdgcn_wmma_f32_16x16x32_bf16(
                    false, a, false, bb, (short)0, acc[c], false, false);
            }
        }
#pragma unroll
        for (int c = 0; c < 6; ++c) {
            int col = (w * 6 + c) * 16 + n;
            float bv = bias[col];
#pragma unroll
            for (int g = 0; g < 8; ++g) {
                int row = row0 + rt * 16 + half * 8 + g;   // row = b*T + t
                int b = row >> 9, t = row & (TT - 1);
                xg[((size_t)t * BB + b) * GG + col] = acc[c][g] + bv;
            }
        }
    }
}

// ---------------------------------------------------------------------------
// Kernel 3: persistent GRU recurrence. Block = 16 batch rows, 8 waves.
// Wave w owns gate columns [32w,32w+32): its 6 WMMA col tiles are the matching
// r/z/n tiles, so gate math is fully in-register. h (f32) lives in VGPRs in
// D-layout; bf16 A-fragments of h live in 8KB LDS. The 48KB xg panel for the
// current step is DMA'd global->LDS with ASYNC loads overlapped with the WMMAs.
// ---------------------------------------------------------------------------
__global__ __launch_bounds__(256)
void gru_recur(const float* __restrict__ xg,
               const unsigned short* __restrict__ whh,
               const float* __restrict__ bhh,
               unsigned short* __restrict__ h_out_bf16,  // [B,T,H] or null
               float* __restrict__ h_last) {             // [B,H]   or null
    __shared__ __align__(32) unsigned short hfrag[8 * 32 * 16];  //  8 KB
    __shared__ __align__(16) float xpanel[16 * GG];              // 48 KB
    const int tid = threadIdx.x;
    const int b0  = blockIdx.x * 16;
    const int w = tid >> 5, lane = tid & 31;
    const int half = lane >> 4, n = lane & 15;

    for (int i = tid; i < 4096; i += 256) hfrag[i] = 0;  // h(-1) = 0
    float hprev[16];
#pragma unroll
    for (int i = 0; i < 16; ++i) hprev[i] = 0.f;

    const int ctb[6] = { 2*w, 2*w + 1, 16 + 2*w, 16 + 2*w + 1, 32 + 2*w, 32 + 2*w + 1 };
    const char* wb[6];
#pragma unroll
    for (int c = 0; c < 6; ++c)
        wb[c] = (const char*)whh + (size_t)ctb[c] * 8192 + lane * 32;   // + kt*1024
    const char* hb = (const char*)hfrag + lane * 32;                     // + kt*1024

    // async-staging bases: LDS offset = low 32 bits of generic shared address
    const unsigned lds0 = (unsigned)(unsigned long long)(const void*)&xpanel[0] + tid * 16;
    const char* g0 = (const char*)xg + (size_t)b0 * GG * 4 + tid * 16;

    __syncthreads();

    for (int t = 0; t < TT; ++t) {
        // ---- issue async DMA of this step's xg panel (overlaps the WMMAs) ----
        {
            const char* g = g0 + (size_t)t * (BB * GG * 4);
#pragma unroll
            for (int i = 0; i < 12; ++i) {      // 256 thr * 12 * 16B = 48 KB
                unsigned la = lds0 + i * 4096;
                unsigned long long ga = (unsigned long long)(g + i * 4096);
                asm volatile("global_load_async_to_lds_b128 %0, %1, off"
                             :: "v"(la), "v"(ga) : "memory");
            }
        }

        // ---- hg = h @ Whh^T (bf16 WMMA, K=256) ----
        v8f acc[6] = {};
#pragma unroll
        for (int kt = 0; kt < 8; ++kt) {
            v16bf a = *(const v16bf*)(hb + kt * 1024);
#pragma unroll
            for (int c = 0; c < 6; ++c) {
                v16bf bb = *(const v16bf*)(wb[c] + kt * 1024);
                acc[c] = __builtin_amdgcn_wmma_f32_16x16x32_bf16(
                    false, a, false, bb, (short)0, acc[c], false, false);
            }
        }

        asm volatile("s_wait_asynccnt 0x0" ::: "memory");  // my panel chunks landed
        __syncthreads();   // everyone's chunks landed + all A-frag reads done

        // ---- fused gates, in-register ----
#pragma unroll
        for (int tt = 0; tt < 2; ++tt) {
            const int j  = 32 * w + 16 * tt + n;
            const float br = bhh[j], bz = bhh[HH + j], bn = bhh[2 * HH + j];
            const int kt = j >> 5, kk = j & 31;
            int ha, e; a_map(kk, ha, e);
#pragma unroll
            for (int g = 0; g < 8; ++g) {
                const int m = half * 8 + g;                 // batch row in tile
                const int xrow = m * GG;
                float xr = xpanel[xrow + j];
                float xz = xpanel[xrow + HH + j];
                float xn = xpanel[xrow + 2 * HH + j];
                float hr = acc[tt][g] + br;
                float hz = acc[2 + tt][g] + bz;
                float hn = acc[4 + tt][g] + bn;
                float r  = 1.f / (1.f + __expf(-(xr + hr)));
                float z  = 1.f / (1.f + __expf(-(xz + hz)));
                float nn = tanhf(xn + r * hn);
                const int idx = tt * 8 + g;
                float hnew = (1.f - z) * nn + z * hprev[idx];
                hprev[idx] = hnew;
                unsigned short us = f2bf(hnew);
                hfrag[(kt * 32 + (ha * 16 + m)) * 16 + e] = us;   // next-step A frag
                if (h_out_bf16)
                    h_out_bf16[((size_t)(b0 + m) * TT + t) * HH + j] = us;
                if (h_last && t == TT - 1)
                    h_last[(b0 + m) * HH + j] = hnew;
            }
        }
        __syncthreads();   // h published (and panel reads done) before next step
    }
}

// ---------------------------------------------------------------------------
// Kernel 4: out[b,o] = relu(h_last[b,:]) . fc_w[o,:] + fc_b[o]   (128x2)
// ---------------------------------------------------------------------------
__global__ void gru_fc(const float* __restrict__ hlast,
                       const float* __restrict__ fw,
                       const float* __restrict__ fb,
                       float* __restrict__ out) {
    int tid = threadIdx.x;
    if (tid >= 256) return;
    int b = tid >> 1, o = tid & 1;
    float s = fb[o];
    for (int j = 0; j < HH; ++j)
        s += fmaxf(hlast[b * HH + j], 0.f) * fw[o * HH + j];
    out[b * 2 + o] = s;
}

// ---------------------------------------------------------------------------
extern "C" void kernel_launch(void* const* d_in, const int* in_sizes, int n_in,
                              void* d_out, int out_size, void* d_ws, size_t ws_size,
                              hipStream_t stream) {
    (void)in_sizes; (void)n_in; (void)out_size; (void)ws_size;
    const float* x     = (const float*)d_in[0];
    const float* w_ih0 = (const float*)d_in[1];
    const float* w_hh0 = (const float*)d_in[2];
    const float* b_ih0 = (const float*)d_in[3];
    const float* b_hh0 = (const float*)d_in[4];
    const float* w_ih1 = (const float*)d_in[5];
    const float* w_hh1 = (const float*)d_in[6];
    const float* b_ih1 = (const float*)d_in[7];
    const float* b_hh1 = (const float*)d_in[8];
    const float* fc_w  = (const float*)d_in[9];
    const float* fc_b  = (const float*)d_in[10];

    char* ws = (char*)d_ws;
    // workspace layout (bytes)
    float*          xg    = (float*)(ws);                          // 128*512*768*4 = 201326592
    unsigned short* h1    = (unsigned short*)(ws + 201326592);     // 128*512*256*2 =  33554432
    unsigned short* wih0p = (unsigned short*)(ws + 234881024);     // 768*64*2  =  98304
    unsigned short* whh0p = (unsigned short*)(ws + 234979328);     // 768*256*2 = 393216
    unsigned short* wih1p = (unsigned short*)(ws + 235372544);     // 393216
    unsigned short* whh1p = (unsigned short*)(ws + 235765760);     // 393216
    float*          hlast = (float*)(ws + 236158976);              // 128*256*4 = 131072

    // 1) pack weights to WMMA B-fragment bf16
    gru_pack_b<<<(GG * II + 255) / 256, 256, 0, stream>>>(w_ih0, wih0p, II);
    gru_pack_b<<<(GG * HH + 255) / 256, 256, 0, stream>>>(w_hh0, whh0p, HH);
    gru_pack_b<<<(GG * HH + 255) / 256, 256, 0, stream>>>(w_ih1, wih1p, HH);
    gru_pack_b<<<(GG * HH + 255) / 256, 256, 0, stream>>>(w_hh1, whh1p, HH);

    // 2) layer 0: xg0 = x @ Wih0^T + b ; recurrence -> h1 (bf16)
    gru_inproj<II, false><<<(BB * TT) / 64, 256, 0, stream>>>(x, wih0p, b_ih0, xg);
    gru_recur<<<BB / 16, 256, 0, stream>>>(xg, whh0p, b_hh0, h1, nullptr);

    // 3) layer 1: xg1 = h1 @ Wih1^T + b ; recurrence -> h_last only
    gru_inproj<HH, true><<<(BB * TT) / 64, 256, 0, stream>>>(h1, wih1p, b_ih1, xg);
    gru_recur<<<BB / 16, 256, 0, stream>>>(xg, whh1p, b_hh1, nullptr, hlast);

    // 4) FC head
    gru_fc<<<1, 256, 0, stream>>>(hlast, fc_w, fc_b, (float*)d_out);
}
